// AgentAwareAttention_61168924230095
// MI455X (gfx1250) — matrix operand order repaired and checked by reading, not compiled, not run on hardware
//
#include <hip/hip_runtime.h>
#include <hip/hip_bf16.h>
#include <math.h>

// ---------------------------------------------------------------------------
// Types for CDNA5 WMMA (wave32). bf16 A/B fragments = 16 elems/lane (8 VGPRs),
// f32 C/D = 8 elems/lane (8 VGPRs).
// ---------------------------------------------------------------------------
typedef __bf16 bf16_t;
typedef __bf16 v16bf __attribute__((ext_vector_type(16)));
typedef float  v8f   __attribute__((ext_vector_type(8)));
typedef unsigned int u32x4 __attribute__((ext_vector_type(4)));

union FragU { u32x4 u[2]; v16bf v; };

// Problem constants (from reference setup)
#define LQ 1024
#define NB 8
#define EE 512
#define HH 8
#define DH 64
#define AGM 15        // num_agent=16 -> mask (l & 15) == (s & 15)
#define SCALE 0.125f  // Dh^-0.5

// --- optional CDNA5 async global->LDS staging ------------------------------
#if defined(__has_builtin)
#if __has_builtin(__builtin_amdgcn_global_load_async_to_lds_b128) && \
    __has_builtin(__builtin_amdgcn_s_wait_asynccnt)
#define USE_ASYNC_LDS 1
#endif
#endif

#ifdef USE_ASYNC_LDS
// Builtin signature (from probe diagnostic): param0 is
// `__vector_size__(16) int __device__*` == int4 in target AS1.
typedef int i32x4 __attribute__((vector_size(16)));
typedef __attribute__((address_space(1))) i32x4* as1p;
typedef __attribute__((address_space(3))) i32x4* as3p;
// integer round-trips: AS1 global addr == generic addr; AS3 offset == low 32
// bits of the generic LDS address (aperture rule).
__device__ __forceinline__ as1p to_as1(const void* p) {
  return (as1p)(unsigned long long)p;
}
__device__ __forceinline__ as3p to_as3(void* p) {
  return (as3p)(unsigned int)(unsigned long long)p;
}
#endif

__device__ __forceinline__ v8f wmma_bf16(v16bf a, v16bf b, v8f c) {
  return __builtin_amdgcn_wmma_f32_16x16x32_bf16(false, a, false, b, (short)0, c,
                                                 false, false);
}

// A fragment (16x32, M x K), row-major source, ld in elements.
__device__ __forceinline__ v16bf load_a_frag(const bf16_t* base, int ld, int row0,
                                             int k0, int lane) {
  const int half = (lane >> 4) & 1, idx = lane & 15;
  const bf16_t* p = base + (long)(row0 + idx) * ld + k0 + half * 8;
  FragU f;
  f.u[0] = *reinterpret_cast<const u32x4*>(p);
  f.u[1] = *reinterpret_cast<const u32x4*>(p + 16);
  return f.v;
}

// B fragment (32x16, K x N) where per-lane data = row `col` of a row-major
// (N x K) matrix.
__device__ __forceinline__ v16bf load_b_frag(const bf16_t* base, int ld, int col0,
                                             int k0, int lane) {
  const int half = (lane >> 4) & 1, idx = lane & 15;
  const bf16_t* p = base + (long)(col0 + idx) * ld + k0 + half * 8;
  FragU f;
  f.u[0] = *reinterpret_cast<const u32x4*>(p);
  f.u[1] = *reinterpret_cast<const u32x4*>(p + 16);
  return f.v;
}

// ---------------------------------------------------------------------------
// f32 -> bf16 convert (grid-stride)
// ---------------------------------------------------------------------------
__global__ void cvt_bf16_kernel(const float* __restrict__ src,
                                bf16_t* __restrict__ dst, int n) {
  for (int i = blockIdx.x * blockDim.x + threadIdx.x; i < n;
       i += gridDim.x * blockDim.x)
    dst[i] = (bf16_t)src[i];
}

// ---------------------------------------------------------------------------
// Fused in-projection: P = X(8192x512) @ Wcat^T(2560x512) + bias, scattered
// into per-head layouts.  q/qs scaled by Dh^-0.5.  v stored transposed.
// Wave computes a 64x64 tile (4x4 WMMA sub-tiles) -> 16 WMMAs per 8 frags.
// Scatter segment/head/d decided once per column sub-tile (uniform there).
// ---------------------------------------------------------------------------
__global__ __launch_bounds__(256) void proj_kernel(
    const bf16_t* __restrict__ X, const bf16_t* __restrict__ Wc,
    const float* __restrict__ bias_qkv, const float* __restrict__ bias_self,
    bf16_t* __restrict__ q, bf16_t* __restrict__ k, bf16_t* __restrict__ vT,
    bf16_t* __restrict__ qs, bf16_t* __restrict__ ks) {
  const int lane = threadIdx.x & 31;
  const int wave = threadIdx.x >> 5;
  const int row0 = blockIdx.x * 128 + (wave >> 2) * 64;
  const int col0 = blockIdx.y * 256 + (wave & 3) * 64;
  const int half = lane >> 4, idx = lane & 15;

  v8f acc[4][4];
#pragma unroll
  for (int i = 0; i < 4; ++i)
#pragma unroll
    for (int j = 0; j < 4; ++j) acc[i][j] = v8f{};

  for (int kk = 0; kk < EE; kk += 32) {
    v16bf a[4], b[4];
#pragma unroll
    for (int i = 0; i < 4; ++i) a[i] = load_a_frag(X, EE, row0 + i * 16, kk, lane);
#pragma unroll
    for (int j = 0; j < 4; ++j) b[j] = load_b_frag(Wc, EE, col0 + j * 16, kk, lane);
#pragma unroll
    for (int i = 0; i < 4; ++i)
#pragma unroll
      for (int j = 0; j < 4; ++j) acc[i][j] = wmma_bf16(a[i], b[j], acc[i][j]);
  }

#pragma unroll
  for (int j = 0; j < 4; ++j) {
    const int cb = col0 + j * 16;              // uniform, 16-aligned
    const int c = cb + idx;
    const float bv = (cb < 3 * EE) ? bias_qkv[c] : bias_self[c - 3 * EE];
    const int seg = cb >> 9;                   // 0..4 (uniform per sub-tile)
    const int cc = cb & 511;
    const int h = cc >> 6;                     // head (uniform)
    const int d = (cc & 63) + idx;             // dim within head

    if (seg == 2) {  // v -> transposed [n][h][d][L]
      bf16_t* base = vT + ((long)h * DH + d) * LQ;
#pragma unroll
      for (int i = 0; i < 4; ++i)
#pragma unroll
        for (int r = 0; r < 8; ++r) {
          const int row = row0 + i * 16 + r + half * 8;  // row = l*NB + n
          const int l = row >> 3, n = row & 7;
          base[(long)n * (HH * DH * LQ) + l] = (bf16_t)(acc[i][j][r] + bv);
        }
    } else {
      bf16_t* dst;
      float scl;
      if (seg == 0)      { dst = q;  scl = SCALE; }
      else if (seg == 1) { dst = k;  scl = 1.f; }
      else if (seg == 3) { dst = qs; scl = SCALE; }
      else               { dst = ks; scl = 1.f; }
      bf16_t* base = dst + (long)h * (LQ * DH) + d;
#pragma unroll
      for (int i = 0; i < 4; ++i)
#pragma unroll
        for (int r = 0; r < 8; ++r) {
          const int row = row0 + i * 16 + r + half * 8;
          const int l = row >> 3, n = row & 7;
          base[(long)n * (HH * LQ * DH) + l * DH] =
              (bf16_t)((acc[i][j][r] + bv) * scl);
        }
    }
  }
}

// ---------------------------------------------------------------------------
// Fused agent-aware flash attention.  One block = one (n,h) pair x 128 query
// rows; 8 waves, each wave owns 16 query rows and streams 32-key chunks.
// K / K_self / V^T chunks staged into LDS (async-to-LDS when available).
// ---------------------------------------------------------------------------
__global__ __launch_bounds__(256) void attn_kernel(
    const bf16_t* __restrict__ q, const bf16_t* __restrict__ k,
    const bf16_t* __restrict__ vT, const bf16_t* __restrict__ qs,
    const bf16_t* __restrict__ ks, const float* __restrict__ amask,
    bf16_t* __restrict__ attnout) {
  __shared__ __align__(16) bf16_t kT[32 * 64];        // keys rows, ld=64
  __shared__ __align__(16) bf16_t sT[32 * 64];        // k_self rows, ld=64
  __shared__ __align__(16) bf16_t vS[64 * 32];        // V^T rows (d), ld=32
  __shared__ __align__(16) bf16_t ldsP[8 * 16 * 32];  // per-wave P tiles

  const int lane = threadIdx.x & 31;
  const int wave = threadIdx.x >> 5;
  const int pair = blockIdx.x >> 3;            // n*H + h
  const int n = pair >> 3, h = pair & 7;
  const int l0 = (blockIdx.x & 7) * 128 + wave * 16;
  const int half = lane >> 4, idx = lane & 15;

  const bf16_t* qn  = q  + (long)pair * LQ * DH;
  const bf16_t* qsn = qs + (long)pair * LQ * DH;
  const bf16_t* kn  = k  + (long)pair * LQ * DH;
  const bf16_t* ksn = ks + (long)pair * LQ * DH;
  const bf16_t* vn  = vT + (long)pair * DH * LQ;

  // Q fragments (Dh=64 -> two K-steps), loaded once.
  const v16bf aq0 = load_a_frag(qn, DH, l0, 0, lane);
  const v16bf aq1 = load_a_frag(qn, DH, l0, 32, lane);
  const v16bf as0 = load_a_frag(qsn, DH, l0, 0, lane);
  const v16bf as1 = load_a_frag(qsn, DH, l0, 32, lane);

  v8f o[4] = {v8f{}, v8f{}, v8f{}, v8f{}};
  float mrun[8], lsum[8];
#pragma unroll
  for (int r = 0; r < 8; ++r) { mrun[r] = -INFINITY; lsum[r] = 0.f; }

  bf16_t* myP = ldsP + wave * (16 * 32);
  const int e8 = threadIdx.x * 8;              // staging element offset
  const int srow = e8 >> 6, scol = e8 & 63;    // for 32x64 tiles
  const int drow = e8 >> 5, dcol = e8 & 31;    // for 64x32 tile

  for (int s0 = 0; s0 < LQ; s0 += 32) {
    // --- cooperative stage of this key chunk into LDS ---
#ifdef USE_ASYNC_LDS
    __builtin_amdgcn_global_load_async_to_lds_b128(
        to_as1(&kn[(s0 + srow) * DH + scol]), to_as3(&kT[e8]), 0, 0);
    __builtin_amdgcn_global_load_async_to_lds_b128(
        to_as1(&ksn[(s0 + srow) * DH + scol]), to_as3(&sT[e8]), 0, 0);
    __builtin_amdgcn_global_load_async_to_lds_b128(
        to_as1(&vn[drow * LQ + s0 + dcol]), to_as3(&vS[e8]), 0, 0);
    __builtin_amdgcn_s_wait_asynccnt(0);
#else
    *reinterpret_cast<u32x4*>(&kT[e8]) =
        *reinterpret_cast<const u32x4*>(&kn[(s0 + srow) * DH + scol]);
    *reinterpret_cast<u32x4*>(&sT[e8]) =
        *reinterpret_cast<const u32x4*>(&ksn[(s0 + srow) * DH + scol]);
    *reinterpret_cast<u32x4*>(&vS[e8]) =
        *reinterpret_cast<const u32x4*>(&vn[drow * LQ + s0 + dcol]);
#endif
    __syncthreads();

    // --- scores: q.k^T and q_self.k_self^T, 16 queries x 32 keys ---
    v8f sc[2], ss[2];
#pragma unroll
    for (int sub = 0; sub < 2; ++sub) {
      v16bf bk0 = load_b_frag(kT, DH, sub * 16, 0, lane);
      v16bf bk1 = load_b_frag(kT, DH, sub * 16, 32, lane);
      v8f a = {};
      a = wmma_bf16(aq0, bk0, a);
      a = wmma_bf16(aq1, bk1, a);
      sc[sub] = a;
      v16bf bs0 = load_b_frag(sT, DH, sub * 16, 0, lane);
      v16bf bs1 = load_b_frag(sT, DH, sub * 16, 32, lane);
      v8f s = {};
      s = wmma_bf16(as0, bs0, s);
      s = wmma_bf16(as1, bs1, s);
      ss[sub] = s;
    }

    // --- agent-mask blend + attn_mask add (element (l = l0+r+8*half, s)) ---
    float p[2][8];
#pragma unroll
    for (int sub = 0; sub < 2; ++sub) {
      const int s = s0 + sub * 16 + idx;
#pragma unroll
      for (int r = 0; r < 8; ++r) {
        const int l = l0 + r + half * 8;
        float v = ((l & AGM) == (s & AGM)) ? ss[sub][r] : sc[sub][r];
        v += amask[l * LQ + s];
        p[sub][r] = v;
      }
    }

    // --- online softmax (row reduce across the 16 key-lanes of each half) ---
#pragma unroll
    for (int r = 0; r < 8; ++r) {
      float cm = fmaxf(p[0][r], p[1][r]);
      cm = fmaxf(cm, __shfl_xor(cm, 1));
      cm = fmaxf(cm, __shfl_xor(cm, 2));
      cm = fmaxf(cm, __shfl_xor(cm, 4));
      cm = fmaxf(cm, __shfl_xor(cm, 8));
      const float mnew = fmaxf(mrun[r], cm);
      const float alpha = __expf(mrun[r] - mnew);
      mrun[r] = mnew;
      const float e0 = __expf(p[0][r] - mnew);
      const float e1 = __expf(p[1][r] - mnew);
      p[0][r] = e0; p[1][r] = e1;
      float rs = e0 + e1;
      rs += __shfl_xor(rs, 1);
      rs += __shfl_xor(rs, 2);
      rs += __shfl_xor(rs, 4);
      rs += __shfl_xor(rs, 8);
      lsum[r] = lsum[r] * alpha + rs;
      o[0][r] *= alpha; o[1][r] *= alpha; o[2][r] *= alpha; o[3][r] *= alpha;
    }

    // --- relayout P (C layout -> A layout) through per-wave LDS tile ---
#pragma unroll
    for (int sub = 0; sub < 2; ++sub)
#pragma unroll
      for (int r = 0; r < 8; ++r)
        myP[(r + half * 8) * 32 + sub * 16 + idx] = (bf16_t)p[sub][r];
    // same-wave DS ops are in order; only this wave touches myP
    const v16bf pa = load_a_frag(myP, 32, 0, 0, lane);

    // --- O += P(16x32) . V(32x64); V^T rows contiguous over keys in LDS ---
#pragma unroll
    for (int t = 0; t < 4; ++t) {
      v16bf bv = load_b_frag(vS, 32, t * 16, 0, lane);
      o[t] = wmma_bf16(pa, bv, o[t]);
    }
    __syncthreads();   // protect kT/sT/vS before next chunk's staging
  }

  // --- normalize, store to (L*N, E) bf16 for the out-projection GEMM ---
#pragma unroll
  for (int r = 0; r < 8; ++r) {
    const int l = l0 + r + half * 8;
    const float inv = 1.0f / lsum[r];
    bf16_t* row = attnout + ((long)l * NB + n) * EE + h * DH;
    row[0 * 16 + idx] = (bf16_t)(o[0][r] * inv);
    row[1 * 16 + idx] = (bf16_t)(o[1][r] * inv);
    row[2 * 16 + idx] = (bf16_t)(o[2][r] * inv);
    row[3 * 16 + idx] = (bf16_t)(o[3][r] * inv);
  }
}

// ---------------------------------------------------------------------------
// Out-projection: out = attn(8192x512) @ Wout^T(512x512) + bias, f32 out.
// ---------------------------------------------------------------------------
__global__ __launch_bounds__(256) void outproj_kernel(
    const bf16_t* __restrict__ A, const bf16_t* __restrict__ W,
    const float* __restrict__ bias, float* __restrict__ out) {
  const int lane = threadIdx.x & 31;
  const int wave = threadIdx.x >> 5;
  const int row0 = blockIdx.x * 128 + (wave >> 2) * 64;
  const int col0 = blockIdx.y * 256 + (wave & 3) * 64;
  const int half = lane >> 4, idx = lane & 15;

  v8f acc[4][4];
#pragma unroll
  for (int i = 0; i < 4; ++i)
#pragma unroll
    for (int j = 0; j < 4; ++j) acc[i][j] = v8f{};

  for (int kk = 0; kk < EE; kk += 32) {
    v16bf a[4], b[4];
#pragma unroll
    for (int i = 0; i < 4; ++i) a[i] = load_a_frag(A, EE, row0 + i * 16, kk, lane);
#pragma unroll
    for (int j = 0; j < 4; ++j) b[j] = load_b_frag(W, EE, col0 + j * 16, kk, lane);
#pragma unroll
    for (int i = 0; i < 4; ++i)
#pragma unroll
      for (int j = 0; j < 4; ++j) acc[i][j] = wmma_bf16(a[i], b[j], acc[i][j]);
  }

#pragma unroll
  for (int j = 0; j < 4; ++j) {
    const int c = col0 + j * 16 + idx;
    const float bv = bias[c];
#pragma unroll
    for (int i = 0; i < 4; ++i)
#pragma unroll
      for (int r = 0; r < 8; ++r)
        out[(long)(row0 + i * 16 + r + half * 8) * EE + c] = acc[i][j][r] + bv;
  }
}

// ---------------------------------------------------------------------------
extern "C" void kernel_launch(void* const* d_in, const int* in_sizes, int n_in,
                              void* d_out, int out_size, void* d_ws, size_t ws_size,
                              hipStream_t stream) {
  const float* query = (const float*)d_in[0];   // (L,N,E)
  const float* Wqkv  = (const float*)d_in[1];   // (3E,E)
  const float* bqkv  = (const float*)d_in[2];   // (3E,)
  const float* Wself = (const float*)d_in[3];   // (2E,E)
  const float* bself = (const float*)d_in[4];   // (2E,)
  const float* Wout  = (const float*)d_in[5];   // (E,E)
  const float* bout  = (const float*)d_in[6];   // (E,)
  const float* amask = (const float*)d_in[7];   // (L,L)
  (void)in_sizes; (void)n_in; (void)out_size; (void)ws_size;

  const int M      = LQ * NB;        // 8192 rows
  const int nX     = M * EE;         // 4,194,304
  const int nWqkv  = 3 * EE * EE;    //   786,432
  const int nWself = 2 * EE * EE;    //   524,288
  const int nWout  = EE * EE;        //   262,144
  const int nHead  = NB * HH * LQ * DH;  // 4,194,304 per tensor

  bf16_t* Xbf   = (bf16_t*)d_ws;
  bf16_t* Wcat  = Xbf + nX;            // rows: [Wqkv ; Wself] = 2560 x 512
  bf16_t* Woutb = Wcat + (nWqkv + nWself);
  bf16_t* qb    = Woutb + nWout;
  bf16_t* kb    = qb  + nHead;
  bf16_t* vTb   = kb  + nHead;
  bf16_t* qsb   = vTb + nHead;
  bf16_t* ksb   = qsb + nHead;
  bf16_t* attnb = ksb + nHead;

  cvt_bf16_kernel<<<2048, 256, 0, stream>>>(query, Xbf, nX);
  cvt_bf16_kernel<<<1024, 256, 0, stream>>>(Wqkv, Wcat, nWqkv);
  cvt_bf16_kernel<<<1024, 256, 0, stream>>>(Wself, Wcat + nWqkv, nWself);
  cvt_bf16_kernel<<<512, 256, 0, stream>>>(Wout, Woutb, nWout);

  // 8192x2560 output: block tile 128x256 -> grid (64, 10)
  proj_kernel<<<dim3(M / 128, 10), 256, 0, stream>>>(Xbf, Wcat, bqkv, bself,
                                                     qb, kb, vTb, qsb, ksb);

  // 64 (n,h) pairs x 8 query blocks of 128 rows
  attn_kernel<<<dim3(NB * HH * 8), 256, 0, stream>>>(qb, kb, vTb, qsb, ksb,
                                                     amask, attnb);

  // 8192x512 output: block tile 128x256 -> grid (64, 2)
  outproj_kernel<<<dim3(M / 128, 2), 256, 0, stream>>>(attnb, Woutb, bout,
                                                       (float*)d_out);
}
